// PathWalkModel_23244363006605
// MI455X (gfx1250) — matrix operand order repaired and checked by reading, not compiled, not run on hardware
//
#include <hip/hip_runtime.h>
#include <hip/hip_bf16.h>

// ---------------------------------------------------------------------------
// PathWalkModel for MI455X (gfx1250, wave32, WMMA f32<-f16).
// B=32, S=64, K=32, E=256, V=100000, NC=2.
// ---------------------------------------------------------------------------

typedef __attribute__((ext_vector_type(16))) _Float16 v16h;
typedef __attribute__((ext_vector_type(8)))  _Float16 v8h;
typedef __attribute__((ext_vector_type(8)))  float    v8f;

constexpr int Bn = 32, Sn = 64, Kn = 32, En = 256;
constexpr int BSn = Bn * Sn;          // 2048
constexpr int G4E = 4 * En;           // 1024

// ---- WMMA helpers ----------------------------------------------------------

__device__ __forceinline__ v8f wmma_f16(v16h a, v16h b, v8f c) {
  return __builtin_amdgcn_wmma_f32_16x16x32_f16(false, a, false, b, (short)0, c,
                                                false, false);
}

// A fragment: 16x32 f16, row-major source with leading dim ld (halfs).
// Per ISA 7.12.2: lane l holds row (l&15); per-lane halfs are two contiguous
// 8-half runs at k = 8*(l>>4) and k = 8*(l>>4)+16.
__device__ __forceinline__ v16h frag_a(const _Float16* __restrict__ p, int ld) {
  const int lane = threadIdx.x & 31;
  const _Float16* q = p + (lane & 15) * ld + (lane >> 4) * 8;
  v8h lo = *(const v8h*)(q);
  v8h hi = *(const v8h*)(q + 16);
  return __builtin_shufflevector(lo, hi, 0, 1, 2, 3, 4, 5, 6, 7,
                                 8, 9, 10, 11, 12, 13, 14, 15);
}

// B fragment: 32x16 f16 (K x N) taken from weight stored row-major [N,K]:
// lane l holds column (l&15); halfs are two 8-half runs at k = 8*(l>>4) and k = 8*(l>>4)+16.
__device__ __forceinline__ v16h frag_b(const _Float16* __restrict__ p, int ld) {
  const int lane = threadIdx.x & 31;
  const _Float16* q = p + (lane & 15) * ld + (lane >> 4) * 8;
  v8h lo = *(const v8h*)(q);
  v8h hi = *(const v8h*)(q + 16);
  return __builtin_shufflevector(lo, hi, 0, 1, 2, 3, 4, 5, 6, 7,
                                 8, 9, 10, 11, 12, 13, 14, 15);
}

__device__ __forceinline__ float sigmoidf_(float x) {
  return 1.f / (1.f + __expf(-x));
}
__device__ __forceinline__ float gelu_(float x) {
  float t = 0.7978845608f * (x + 0.044715f * x * x * x);
  return 0.5f * x * (1.f + tanhf(t));
}

// ---- small prep kernels ----------------------------------------------------

__global__ void k_cast_f16(const float* __restrict__ s, _Float16* __restrict__ d,
                           int n) {
  int i = blockIdx.x * blockDim.x + threadIdx.x;
  if (i < n) d[i] = (_Float16)s[i];
}

__global__ void k_addv(const float* __restrict__ a, const float* __restrict__ b,
                       float* __restrict__ o, int n) {
  int i = blockIdx.x * blockDim.x + threadIdx.x;
  if (i < n) o[i] = a[i] + b[i];
}

// v[e] = sum_j a2_W[0, 256+j] * a1_W[j, e]   (AttentionLayer collapsed)
__global__ void k_vvec(const float* __restrict__ a1W,
                       const float* __restrict__ a2W, float* __restrict__ v) {
  int e = threadIdx.x;
  float acc = 0.f;
  for (int j = 0; j < En; ++j) acc += a2W[En + j] * a1W[j * En + e];
  v[e] = acc;
}

// gather embedding rows -> f16 destination (optionally t-major permute)
__global__ void k_gather_f16(const float* __restrict__ emb,
                             const int* __restrict__ idx,
                             _Float16* __restrict__ dst, int ld, int coloff,
                             int tmajor) {
  int row = blockIdx.x;
  int j = threadIdx.x;  // 0..255
  int id = idx[row];
  int drow = row;
  if (tmajor) { int b = row / Sn, s = row % Sn; drow = s * Bn + b; }
  dst[(size_t)drow * ld + coloff + j] = (_Float16)emb[(size_t)id * En + j];
}

// ---- Gx = x_emb @ Wih^T + (bih+bhh) ; t-major rows [S*B, 1024] -------------

__global__ __launch_bounds__(512) void k_gemm_gates(
    const _Float16* __restrict__ A,   // [2048,256] f16 (t-major)
    const _Float16* __restrict__ W,   // [1024,256] f16
    const float* __restrict__ bias,   // [1024]
    float* __restrict__ C) {          // [2048,1024] f32
  __shared__ _Float16 As[32 * 256];
  const int r0 = blockIdx.x * 32;
  const int n0 = blockIdx.y * 256;
  {
    const v8h* src = (const v8h*)(A + (size_t)r0 * 256);
    v8h* dv = (v8h*)As;
    for (int i = threadIdx.x; i < 32 * 256 / 8; i += blockDim.x) dv[i] = src[i];
  }
  __syncthreads();
  const int w = threadIdx.x >> 5, lane = threadIdx.x & 31;
  const int nt = w, half = lane >> 4, cn = lane & 15;
  v8f acc0 = {}, acc1 = {};
#pragma unroll
  for (int kt = 0; kt < 8; ++kt) {
    v16h bf = frag_b(W + (size_t)(n0 + nt * 16) * 256 + kt * 32, 256);
    v16h a0 = frag_a(As + 0 * 16 * 256 + kt * 32, 256);
    v16h a1 = frag_a(As + 1 * 16 * 256 + kt * 32, 256);
    acc0 = wmma_f16(a0, bf, acc0);
    acc1 = wmma_f16(a1, bf, acc1);
  }
  const int col = n0 + nt * 16 + cn;
  const float bv = bias[col];
  const int rb = r0 + half * 8;
#pragma unroll
  for (int r = 0; r < 8; ++r) {
    C[(size_t)(rb + r) * G4E + col] = acc0[r] + bv;
    C[(size_t)(rb + 16 + r) * G4E + col] = acc1[r] + bv;
  }
}

// ---- BiLSTM recurrence: 2 blocks (fwd/bwd), 16 waves each ------------------
// Each wave owns hidden n-tile hn; its 4 gate tiles (i,f,g,o) and c-state stay
// in registers; h is exchanged via a 16KB f16 LDS tile per step.

__global__ __launch_bounds__(512) void k_lstm_rec(
    const float* __restrict__ GxF, const float* __restrict__ GxB,
    const _Float16* __restrict__ WhhF, const _Float16* __restrict__ WhhB,
    float* __restrict__ outp) {  // [B,S,512] f32
  __shared__ _Float16 h16[32 * 256];
  const int dir = blockIdx.x;
  const float* __restrict__ Gx = dir ? GxB : GxF;
  const _Float16* __restrict__ Whh = dir ? WhhB : WhhF;
  const int w = threadIdx.x >> 5, lane = threadIdx.x & 31;
  const int hn = w, half = lane >> 4, cn = lane & 15;
  for (int i = threadIdx.x; i < 32 * 256; i += blockDim.x) h16[i] = (_Float16)0.f;
  v8f c[2] = {{}, {}};
  __syncthreads();
  for (int step = 0; step < Sn; ++step) {
    const int t = dir ? (Sn - 1 - step) : step;
    v8f acc[2][4] = {{{}, {}, {}, {}}, {{}, {}, {}, {}}};
#pragma unroll
    for (int kt = 0; kt < 8; ++kt) {
      v16h a0 = frag_a(h16 + 0 * 16 * 256 + kt * 32, 256);
      v16h a1 = frag_a(h16 + 1 * 16 * 256 + kt * 32, 256);
#pragma unroll
      for (int q = 0; q < 4; ++q) {
        v16h bf = frag_b(Whh + (size_t)(q * 256 + hn * 16) * 256 + kt * 32, 256);
        acc[0][q] = wmma_f16(a0, bf, acc[0][q]);
        acc[1][q] = wmma_f16(a1, bf, acc[1][q]);
      }
    }
    __syncthreads();  // all h16 reads done before overwrite
    const int gcol = hn * 16 + cn;
#pragma unroll
    for (int mt = 0; mt < 2; ++mt) {
#pragma unroll
      for (int r = 0; r < 8; ++r) {
        const int m = mt * 16 + half * 8 + r;  // batch row
        const size_t grow = ((size_t)t * Bn + m) * G4E;
        float gi = acc[mt][0][r] + Gx[grow + 0 * 256 + gcol];
        float gf = acc[mt][1][r] + Gx[grow + 1 * 256 + gcol];
        float gg = acc[mt][2][r] + Gx[grow + 2 * 256 + gcol];
        float go = acc[mt][3][r] + Gx[grow + 3 * 256 + gcol];
        float cv = sigmoidf_(gf) * c[mt][r] + sigmoidf_(gi) * tanhf(gg);
        c[mt][r] = cv;
        float hv = sigmoidf_(go) * tanhf(cv);
        outp[((size_t)m * Sn + t) * 512 + dir * 256 + gcol] = hv;
        h16[m * 256 + gcol] = (_Float16)hv;
      }
    }
    __syncthreads();  // h16 visible to all waves for next step
  }
}

// ---- sub + concat builders -------------------------------------------------

__global__ void k_subcat(const float* __restrict__ outp,
                         _Float16* __restrict__ catIE,
                         _Float16* __restrict__ catNE) {
  const int bs = blockIdx.x, s = bs % Sn;
  const int j = threadIdx.x;  // 0..255
  const float* row = outp + (size_t)bs * 512;
  float o0 = row[j], o1 = row[256 + j];
  float s0 = 0.f, s1 = 0.f;
  if (s < Sn - 1) { s0 = row[512 + j] - o0; s1 = row[768 + j] - o1; }
  catIE[(size_t)bs * 768 + 256 + j] = (_Float16)s0;
  catIE[(size_t)bs * 768 + 512 + j] = (_Float16)s1;
  catNE[(size_t)bs * 768 + j] = (_Float16)o0;
  catNE[(size_t)bs * 768 + 256 + j] = (_Float16)o1;
}

// ---- fused Linear(768->256) + LayerNorm + GELU -----------------------------

__global__ __launch_bounds__(512) void k_fuse(
    const _Float16* __restrict__ A,  // [2048,768] f16
    const _Float16* __restrict__ W,  // [256,768] f16
    const float* __restrict__ bias, const float* __restrict__ gam,
    const float* __restrict__ bet, _Float16* __restrict__ out16, int ld16,
    int coloff16, float* __restrict__ out32) {
  __shared__ _Float16 As[32 * 768];
  __shared__ float sS[32], sQ[32], sMu[32], sRs[32];
  const int r0 = blockIdx.x * 32;
  {
    const v8h* src = (const v8h*)(A + (size_t)r0 * 768);
    v8h* dv = (v8h*)As;
    for (int i = threadIdx.x; i < 32 * 768 / 8; i += blockDim.x) dv[i] = src[i];
  }
  if (threadIdx.x < 32) { sS[threadIdx.x] = 0.f; sQ[threadIdx.x] = 0.f; }
  __syncthreads();
  const int w = threadIdx.x >> 5, lane = threadIdx.x & 31;
  const int nt = w, half = lane >> 4, cn = lane & 15;
  v8f acc[2] = {{}, {}};
  for (int kt = 0; kt < 24; ++kt) {
    v16h bf = frag_b(W + (size_t)(nt * 16) * 768 + kt * 32, 768);
#pragma unroll
    for (int mt = 0; mt < 2; ++mt) {
      v16h af = frag_a(As + mt * 16 * 768 + kt * 32, 768);
      acc[mt] = wmma_f16(af, bf, acc[mt]);
    }
  }
  const int col = nt * 16 + cn;
  const float bv = bias[col];
#pragma unroll
  for (int mt = 0; mt < 2; ++mt) {
#pragma unroll
    for (int r = 0; r < 8; ++r) {
      float v = acc[mt][r] + bv;
      acc[mt][r] = v;
      float s = v, q = v * v;
      s += __shfl_xor(s, 1); s += __shfl_xor(s, 2);
      s += __shfl_xor(s, 4); s += __shfl_xor(s, 8);
      q += __shfl_xor(q, 1); q += __shfl_xor(q, 2);
      q += __shfl_xor(q, 4); q += __shfl_xor(q, 8);
      if (cn == 0) {
        int row = mt * 16 + half * 8 + r;
        atomicAdd(&sS[row], s);
        atomicAdd(&sQ[row], q);
      }
    }
  }
  __syncthreads();
  if (threadIdx.x < 32) {
    float mu = sS[threadIdx.x] * (1.f / 256.f);
    float var = sQ[threadIdx.x] * (1.f / 256.f) - mu * mu;
    sMu[threadIdx.x] = mu;
    sRs[threadIdx.x] = rsqrtf(var + 1e-5f);
  }
  __syncthreads();
  const float gv = gam[col], btv = bet[col];
#pragma unroll
  for (int mt = 0; mt < 2; ++mt) {
#pragma unroll
    for (int r = 0; r < 8; ++r) {
      int row = mt * 16 + half * 8 + r;
      float hh = (acc[mt][r] - sMu[row]) * sRs[row] * gv + btv;
      float gl = gelu_(hh);
      size_t grow = (size_t)(r0 + row);
      if (out16) out16[grow * ld16 + coloff16 + col] = (_Float16)gl;
      if (out32) out32[grow * 256 + col] = gl;
    }
  }
}

// ---- neighborhood: gather + fuse + collapsed attention, one (b,s)/block ----

__global__ __launch_bounds__(512) void k_nbhd(
    const float* __restrict__ emb, const int* __restrict__ e1,
    const int* __restrict__ on, const int* __restrict__ e2,
    const _Float16* __restrict__ W, const float* __restrict__ bias,
    const float* __restrict__ gam, const float* __restrict__ bet,
    const float* __restrict__ vvec, float* __restrict__ x_attn) {  // [2048,256]
  __shared__ _Float16 As[32 * 768];
  __shared__ float sS[32], sQ[32], sMu[32], sRs[32], sDot[32], sProb[32];
  const int bs = blockIdx.x;
  // gather: rows are the K=32 neighbors; cols [e1|on|e2]
  for (int chunk = threadIdx.x; chunk < 32 * 3 * 64; chunk += blockDim.x) {
    int row = chunk / 192, rem = chunk % 192;
    int part = rem / 64, i4 = rem % 64;
    const int* idx = (part == 0) ? e1 : ((part == 1) ? on : e2);
    int id = idx[bs * Kn + row];
    float4 f = ((const float4*)(emb + (size_t)id * En))[i4];
    _Float16* d = As + row * 768 + part * 256 + i4 * 4;
    d[0] = (_Float16)f.x; d[1] = (_Float16)f.y;
    d[2] = (_Float16)f.z; d[3] = (_Float16)f.w;
  }
  if (threadIdx.x < 32) {
    sS[threadIdx.x] = 0.f; sQ[threadIdx.x] = 0.f; sDot[threadIdx.x] = 0.f;
  }
  __syncthreads();
  const int w = threadIdx.x >> 5, lane = threadIdx.x & 31;
  const int nt = w, half = lane >> 4, cn = lane & 15;
  v8f acc[2] = {{}, {}};
  for (int kt = 0; kt < 24; ++kt) {
    v16h bf = frag_b(W + (size_t)(nt * 16) * 768 + kt * 32, 768);
#pragma unroll
    for (int mt = 0; mt < 2; ++mt) {
      v16h af = frag_a(As + mt * 16 * 768 + kt * 32, 768);
      acc[mt] = wmma_f16(af, bf, acc[mt]);
    }
  }
  const int col = nt * 16 + cn;
  const float bv = bias[col];
#pragma unroll
  for (int mt = 0; mt < 2; ++mt) {
#pragma unroll
    for (int r = 0; r < 8; ++r) {
      float v = acc[mt][r] + bv;
      acc[mt][r] = v;
      float s = v, q = v * v;
      s += __shfl_xor(s, 1); s += __shfl_xor(s, 2);
      s += __shfl_xor(s, 4); s += __shfl_xor(s, 8);
      q += __shfl_xor(q, 1); q += __shfl_xor(q, 2);
      q += __shfl_xor(q, 4); q += __shfl_xor(q, 8);
      if (cn == 0) {
        int row = mt * 16 + half * 8 + r;
        atomicAdd(&sS[row], s);
        atomicAdd(&sQ[row], q);
      }
    }
  }
  __syncthreads();
  if (threadIdx.x < 32) {
    float mu = sS[threadIdx.x] * (1.f / 256.f);
    float var = sQ[threadIdx.x] * (1.f / 256.f) - mu * mu;
    sMu[threadIdx.x] = mu;
    sRs[threadIdx.x] = rsqrtf(var + 1e-5f);
  }
  __syncthreads();
  const float gv = gam[col], btv = bet[col], vv = vvec[col];
  float nb[2][8];
#pragma unroll
  for (int mt = 0; mt < 2; ++mt) {
#pragma unroll
    for (int r = 0; r < 8; ++r) {
      int row = mt * 16 + half * 8 + r;
      float hh = (acc[mt][r] - sMu[row]) * sRs[row] * gv + btv;
      float gl = gelu_(hh);
      nb[mt][r] = gl;
      float d = gl * vv;  // logit contribution (constant terms cancel in softmax)
      d += __shfl_xor(d, 1); d += __shfl_xor(d, 2);
      d += __shfl_xor(d, 4); d += __shfl_xor(d, 8);
      if (cn == 0) atomicAdd(&sDot[row], d);
    }
  }
  __syncthreads();
  if (w == 0) {  // softmax over K=32 neighbors, one wave
    float x = sDot[lane], mx = x;
    for (int off = 16; off; off >>= 1) mx = fmaxf(mx, __shfl_xor(mx, off));
    float e = __expf(x - mx), sm = e;
    for (int off = 16; off; off >>= 1) sm += __shfl_xor(sm, off);
    sProb[lane] = e / sm;
  }
  __syncthreads();
  float t = 0.f;
#pragma unroll
  for (int mt = 0; mt < 2; ++mt)
#pragma unroll
    for (int r = 0; r < 8; ++r) t += sProb[mt * 16 + half * 8 + r] * nb[mt][r];
  t += __shfl_xor(t, 16);
  if (half == 0) x_attn[(size_t)bs * En + col] = t;
}

// ---- pooling + head --------------------------------------------------------

__global__ __launch_bounds__(256) void k_final(
    const float* __restrict__ xsum, const float* __restrict__ x_attn,
    const float* __restrict__ x_mask, const float* __restrict__ saW,
    const float* __restrict__ saB, const float* __restrict__ encW,
    const float* __restrict__ prW, const float* __restrict__ prB,
    float* __restrict__ outp) {
  __shared__ float sc[64], aw[64], sent[256], nbh[256];
  const int b = blockIdx.x, tx = threadIdx.x;
  if (tx < 64) {
    const float* r = xsum + ((size_t)b * Sn + tx) * En;
    float a = saB[0];
    for (int n = 0; n < En; ++n) a += r[n] * saW[n];
    sc[tx] = (x_mask[b * Sn + tx] > 0.f) ? a : -1e9f;
  }
  __syncthreads();
  if (tx == 0) {
    float mx = sc[0];
    for (int t = 1; t < 64; ++t) mx = fmaxf(mx, sc[t]);
    float sm = 0.f;
    for (int t = 0; t < 64; ++t) { sc[t] = __expf(sc[t] - mx); sm += sc[t]; }
    for (int t = 0; t < 64; ++t) sc[t] /= sm;
  }
  __syncthreads();
  {
    float a = 0.f;
    for (int t = 0; t < 64; ++t) a += sc[t] * xsum[((size_t)b * Sn + t) * En + tx];
    sent[tx] = a;
  }
  if (tx < 64) {  // hybrid logits: sent & enc_b terms cancel under softmax_t
    const float* r = x_attn + ((size_t)b * Sn + tx) * En;
    float a = 0.f;
    for (int n = 0; n < En; ++n) a += r[n] * encW[En + n];
    aw[tx] = a;
  }
  __syncthreads();
  if (tx == 0) {
    float mx = aw[0];
    for (int t = 1; t < 64; ++t) mx = fmaxf(mx, aw[t]);
    float sm = 0.f;
    for (int t = 0; t < 64; ++t) { aw[t] = __expf(aw[t] - mx); sm += aw[t]; }
    for (int t = 0; t < 64; ++t) aw[t] /= sm;
  }
  __syncthreads();
  {
    float a = 0.f;
    for (int t = 0; t < 64; ++t) a += aw[t] * x_attn[((size_t)b * Sn + t) * En + tx];
    nbh[tx] = a;
  }
  __syncthreads();
  if (tx == 0) {
    float r0 = prB[0], r1 = prB[1];
    for (int n = 0; n < En; ++n) {
      r0 += sent[n] * prW[n] + nbh[n] * prW[En + n];
      r1 += sent[n] * prW[512 + n] + nbh[n] * prW[512 + En + n];
    }
    float mx = fmaxf(r0, r1);
    float e0 = __expf(r0 - mx), e1 = __expf(r1 - mx);
    float s = e0 + e1;
    outp[b * 2 + 0] = e0 / s;
    outp[b * 2 + 1] = e1 / s;
  }
}

// ---------------------------------------------------------------------------

extern "C" void kernel_launch(void* const* d_in, const int* in_sizes, int n_in,
                              void* d_out, int out_size, void* d_ws,
                              size_t ws_size, hipStream_t stream) {
  (void)in_sizes; (void)n_in; (void)out_size; (void)ws_size;

  const int* x = (const int*)d_in[0];
  const float* x_mask = (const float*)d_in[2];
  const int* in_edges = (const int*)d_in[3];
  const int* edge1 = (const int*)d_in[5];
  const int* outern = (const int*)d_in[6];
  const int* edge2 = (const int*)d_in[7];
  const float* embed = (const float*)d_in[9];
  const float* wihF = (const float*)d_in[10];
  const float* whhF = (const float*)d_in[11];
  const float* bihF = (const float*)d_in[12];
  const float* bhhF = (const float*)d_in[13];
  const float* wihB = (const float*)d_in[14];
  const float* whhB = (const float*)d_in[15];
  const float* bihB = (const float*)d_in[16];
  const float* bhhB = (const float*)d_in[17];
  const float* ieW = (const float*)d_in[18];
  const float* ieb = (const float*)d_in[19];
  const float* ieg = (const float*)d_in[20];
  const float* iebt = (const float*)d_in[21];
  const float* neW = (const float*)d_in[22];
  const float* neb = (const float*)d_in[23];
  const float* neg_ = (const float*)d_in[24];
  const float* nebt = (const float*)d_in[25];
  const float* oeW = (const float*)d_in[26];
  const float* oeb = (const float*)d_in[27];
  const float* oeg = (const float*)d_in[28];
  const float* oebt = (const float*)d_in[29];
  const float* a1W = (const float*)d_in[30];
  const float* a2W = (const float*)d_in[32];
  const float* saW = (const float*)d_in[34];
  const float* saB = (const float*)d_in[35];
  const float* encW = (const float*)d_in[36];
  const float* prW = (const float*)d_in[38];
  const float* prB = (const float*)d_in[39];

  // workspace bump allocator (256B aligned)
  char* p = (char*)d_ws;
  auto alloc = [&](size_t bytes) {
    void* r = (void*)p;
    p += (bytes + 255) & ~(size_t)255;
    return r;
  };
  _Float16* wih16F = (_Float16*)alloc(G4E * En * 2);
  _Float16* wih16B = (_Float16*)alloc(G4E * En * 2);
  _Float16* whh16F = (_Float16*)alloc(G4E * En * 2);
  _Float16* whh16B = (_Float16*)alloc(G4E * En * 2);
  _Float16* ieW16 = (_Float16*)alloc(En * 768 * 2);
  _Float16* neW16 = (_Float16*)alloc(En * 768 * 2);
  _Float16* oeW16 = (_Float16*)alloc(En * 768 * 2);
  float* biasF = (float*)alloc(G4E * 4);
  float* biasB = (float*)alloc(G4E * 4);
  float* vvec = (float*)alloc(En * 4);
  _Float16* xemb16 = (_Float16*)alloc((size_t)BSn * En * 2);   // t-major
  _Float16* catIE = (_Float16*)alloc((size_t)BSn * 768 * 2);
  _Float16* catNE = (_Float16*)alloc((size_t)BSn * 768 * 2);
  float* GxF = (float*)alloc((size_t)BSn * G4E * 4);           // t-major
  float* GxB = (float*)alloc((size_t)BSn * G4E * 4);
  float* lstm_out = (float*)alloc((size_t)BSn * 512 * 4);      // [B,S,2E]
  float* xsum = (float*)alloc((size_t)BSn * En * 4);
  float* x_attn = (float*)alloc((size_t)BSn * En * 4);

  // 1) weight prep
  const int nW = G4E * En;
  k_cast_f16<<<(nW + 255) / 256, 256, 0, stream>>>(wihF, wih16F, nW);
  k_cast_f16<<<(nW + 255) / 256, 256, 0, stream>>>(wihB, wih16B, nW);
  k_cast_f16<<<(nW + 255) / 256, 256, 0, stream>>>(whhF, whh16F, nW);
  k_cast_f16<<<(nW + 255) / 256, 256, 0, stream>>>(whhB, whh16B, nW);
  const int nF = En * 768;
  k_cast_f16<<<(nF + 255) / 256, 256, 0, stream>>>(ieW, ieW16, nF);
  k_cast_f16<<<(nF + 255) / 256, 256, 0, stream>>>(neW, neW16, nF);
  k_cast_f16<<<(nF + 255) / 256, 256, 0, stream>>>(oeW, oeW16, nF);
  k_addv<<<(G4E + 255) / 256, 256, 0, stream>>>(bihF, bhhF, biasF, G4E);
  k_addv<<<(G4E + 255) / 256, 256, 0, stream>>>(bihB, bhhB, biasB, G4E);
  k_vvec<<<1, 256, 0, stream>>>(a1W, a2W, vvec);

  // 2) embedding gathers
  k_gather_f16<<<BSn, 256, 0, stream>>>(embed, x, xemb16, 256, 0, 1);
  k_gather_f16<<<BSn, 256, 0, stream>>>(embed, in_edges, catIE, 768, 0, 0);

  // 3) batched LSTM input projection (WMMA), t-major
  k_gemm_gates<<<dim3(BSn / 32, G4E / 256), 512, 0, stream>>>(xemb16, wih16F,
                                                              biasF, GxF);
  k_gemm_gates<<<dim3(BSn / 32, G4E / 256), 512, 0, stream>>>(xemb16, wih16B,
                                                              biasB, GxB);

  // 4) recurrence: fwd + bwd in parallel blocks
  k_lstm_rec<<<2, 512, 0, stream>>>(GxF, GxB, whh16F, whh16B, lstm_out);

  // 5) sub + concat inputs
  k_subcat<<<BSn, 256, 0, stream>>>(lstm_out, catIE, catNE);

  // 6) xe fuse -> catNE[:,512:768] ; xsum fuse -> xsum f32
  k_fuse<<<BSn / 32, 512, 0, stream>>>(catIE, ieW16, ieb, ieg, iebt, catNE,
                                       768, 512, nullptr);
  k_fuse<<<BSn / 32, 512, 0, stream>>>(catNE, neW16, neb, neg_, nebt, nullptr,
                                       0, 0, xsum);

  // 7) neighborhood fuse + collapsed attention -> x_attn
  k_nbhd<<<BSn, 512, 0, stream>>>(embed, edge1, outern, edge2, oeW16, oeb, oeg,
                                  oebt, vvec, x_attn);

  // 8) pooling + head
  k_final<<<Bn, 256, 0, stream>>>(xsum, x_attn, x_mask, saW, saB, encW, prW,
                                  prB, (float*)d_out);
}
